// FCNRZ_72945724555867
// MI455X (gfx1250) — matrix operand order repaired and checked by reading, not compiled, not run on hardware
//
#include <hip/hip_runtime.h>

// ---------------------------------------------------------------------------
// RZ-linear MLP on MI455X (gfx1250, wave32).
// Compute-bound (107 GFLOP, working set ~90MB -> fully L2-resident).
// bf16 WMMA 16x16x32 with 3-term split-fp32 (hi*hi + hi*lo + lo*hi).
// v4: - SPLIT epilogue is a template parameter (kills per-element branch)
//     - gather phase hash/address computed incrementally (n invariant per
//       thread; hash affine in chunk-iteration with step 4*P1)
// Wave tile 32x64 (2 M-frags x 4 N-subtiles), <256 VGPRs, inner loop is
// ds_load_b32 + v_wmma only.
// ---------------------------------------------------------------------------

typedef __attribute__((ext_vector_type(16))) __bf16       v16bf;
typedef __attribute__((ext_vector_type(2)))  __bf16       v2bf;
typedef __attribute__((ext_vector_type(8)))  float        v8f;
typedef __attribute__((ext_vector_type(8)))  unsigned int v8u;
typedef __attribute__((ext_vector_type(4)))  unsigned int v4u;

#define HASH_P1 2654435761u
#define HASH_P2 40503u

// LDS W-tile: 32(K) x 128(N) bf16, two planes (hi, lo), [n][k] order.
// Row stride 36 halfwords; k>=16 stored at k+2 so the two half-wave K-groups
// hit even vs odd dword banks -> zero bank conflicts on the b32 reads.
#define WROW  36
#define PLANE (128 * WROW)   // halfwords per plane

static __device__ __forceinline__ void bf16_split(float f, __bf16& hi, __bf16& lo) {
    hi = (__bf16)f;
    lo = (__bf16)(f - (float)hi);
}

// ---------------------------------------------------------------------------
// Kernel 1: hashed embedding bag + weighted-sum pooling -> split bf16 planes.
// grid = B (2048), block = 256 (one thread per embedding dim d).
// ---------------------------------------------------------------------------
__global__ void rz_embed_pool_kernel(const int* __restrict__ x,
                                     const float* __restrict__ w,
                                     const float* __restrict__ hw_emb,
                                     __bf16* __restrict__ phi,
                                     __bf16* __restrict__ plo) {
    __shared__ int   sx[50];
    __shared__ float sw[50];
    const int b = blockIdx.x;
    const int d = threadIdx.x;            // 0..255
    if (d < 50) {
        sx[d] = x[b * 50 + d];
        sw[d] = w[b * 50 + d];
    }
    __syncthreads();
    const unsigned mask = (1u << 20) - 1u;   // hw_emb size = 2^20
    float acc = 0.0f;
#pragma unroll 5
    for (int t = 0; t < 50; ++t) {
        int xi = sx[t];
        if (xi != 0) {   // uniform across block at each t
            unsigned g = ((unsigned)xi * HASH_P1 + (unsigned)d * HASH_P2 + 1u) & mask;
            acc += sw[t] * hw_emb[g];
        }
    }
    __bf16 h, l;
    bf16_split(acc, h, l);
    phi[(size_t)b * 256 + d] = h;
    plo[(size_t)b * 256 + d] = l;
}

// ---------------------------------------------------------------------------
// Kernel 2: C[M,N] = A[M,K] @ W[K,N],  W[k,n] = hw[(k*P1 + n*P2 + seed) & mask]
// A arrives pre-split (hi/lo bf16 planes, row-major [M][K]).
// Block: 256 thr = 8 waves; block tile 128(M) x 128(N).
// Wave (mq = w>>1, nq = w&1): rows [mq*32, +32), cols [nq*64, +64).
// Per K-chunk of 32: cooperative gather/split of W into LDS planes, then
// 4 N-subtiles x 2 M-frags x 3 x v_wmma_f32_16x16x32_bf16.
// SPLIT: true -> write hi/lo bf16 planes (next layer), false -> write f32.
// ---------------------------------------------------------------------------
template <bool SPLIT>
__global__ void __launch_bounds__(256)
rz_gemm_bf16x3_kernel(const __bf16* __restrict__ Ahi,
                      const __bf16* __restrict__ Alo,
                      const float*  __restrict__ hw,
                      float*        __restrict__ Cf32,
                      __bf16*       __restrict__ Chi,
                      __bf16*       __restrict__ Clo,
                      int M, int N, int K,
                      unsigned seed, unsigned mask) {
    __shared__ __bf16 lds16[2 * PLANE];     // hi plane then lo plane: 18 KB

    const int tid   = threadIdx.x;
    const int wave  = tid >> 5;
    const int lane  = tid & 31;
    const int lhalf = lane >> 4;            // 0: lanes 0-15, 1: lanes 16-31
    const int l16   = lane & 15;
    const int mq    = wave >> 1;            // 0..3 -> 32-row group
    const int nq    = wave & 1;             // 0..1 -> 64-col group

    const int n_base = blockIdx.x * 128;
    const int m_base = blockIdx.y * 128 + mq * 32;

    v8f acc[2][4];
#pragma unroll
    for (int r = 0; r < 2; ++r)
#pragma unroll
        for (int t = 0; t < 4; ++t) acc[r][t] = (v8f){};

    const int khalf = lhalf * 16;           // A K-offset split across halves
    const int kb2   = lhalf * 18;           // B halfword offset (k>=16 -> +2 shim)

    // Gather-phase invariants: this thread always covers column gn, k-pairs
    // k2 = k2b + 2*i (i = 0..7). Hash is affine in i with step 4*P1.
    const int      gn    = tid & 127;       // 0..127 (constant across i)
    const int      k2b   = tid >> 7;        // 0 or 1
    const unsigned hcol  = (unsigned)(n_base + gn) * HASH_P2 + seed;
    const int      lbase = gn * WROW + 2 * k2b;

    for (int k0 = 0; k0 < K; k0 += 32) {
        // ---- cooperative gather + split of W chunk into LDS planes --------
        __syncthreads();                    // previous iteration's reads done
        unsigned hrow = (unsigned)(k0 + 2 * k2b) * HASH_P1 + hcol;
#pragma unroll
        for (int i = 0; i < 8; ++i) {
            unsigned g0 = hrow & mask;
            unsigned g1 = (hrow + HASH_P1) & mask;
            float f0 = hw[g0], f1 = hw[g1];
            __bf16 h0, l0, h1, l1;
            bf16_split(f0, h0, l0);
            bf16_split(f1, h1, l1);
            v2bf ph; ph[0] = h0; ph[1] = h1;
            v2bf pl; pl[0] = l0; pl[1] = l1;
            const int k2   = k2b + 2 * i;
            const int base = lbase + 4 * i + (k2 >= 8 ? 2 : 0);
            *(v2bf*)&lds16[base]         = ph;
            *(v2bf*)&lds16[PLANE + base] = pl;
            hrow += 4u * HASH_P1;
        }
        __syncthreads();

        // ---- A fragments (2 x 16 rows): 16B loads of pre-split planes -----
        // lane layout: vec[0..15] = K khalf..khalf+15
        v16bf a_hi[2], a_lo[2];
#pragma unroll
        for (int r = 0; r < 2; ++r) {
            const size_t abase = (size_t)(m_base + r * 16 + l16) * K + k0 + khalf;
            v4u h0 = *(const v4u*)(Ahi + abase);
            v4u h1 = *(const v4u*)(Ahi + abase + 8);
            v4u l0 = *(const v4u*)(Alo + abase);
            v4u l1 = *(const v4u*)(Alo + abase + 8);
            v8u ha, la;
#pragma unroll
            for (int j = 0; j < 4; ++j) {
                ha[j] = h0[j]; ha[4 + j] = h1[j];
                la[j] = l0[j]; la[4 + j] = l1[j];
            }
            a_hi[r] = __builtin_bit_cast(v16bf, ha);
            a_lo[r] = __builtin_bit_cast(v16bf, la);
        }

        // ---- 4 N-subtiles: ds_load_b32 -> 6 WMMAs each (2 M-frags) --------
#pragma unroll
        for (int t = 0; t < 4; ++t) {
            const int nloc = nq * 64 + t * 16 + l16;
            const __bf16* ph = &lds16[nloc * WROW + kb2];
            const __bf16* pl = ph + PLANE;
            v8u hb, lb;
#pragma unroll
            for (int j = 0; j < 8; ++j) {
                hb[j] = *(const unsigned int*)(ph + 2 * j);
                lb[j] = *(const unsigned int*)(pl + 2 * j);
            }
            v16bf b_hi = __builtin_bit_cast(v16bf, hb);
            v16bf b_lo = __builtin_bit_cast(v16bf, lb);
#pragma unroll
            for (int r = 0; r < 2; ++r) {
                // split-fp32: lo*hi + hi*lo + hi*hi (lo*lo ~2^-16, dropped)
                acc[r][t] = __builtin_amdgcn_wmma_f32_16x16x32_bf16(
                    false, a_lo[r], false, b_hi, (short)0, acc[r][t], false, false);
                acc[r][t] = __builtin_amdgcn_wmma_f32_16x16x32_bf16(
                    false, a_hi[r], false, b_lo, (short)0, acc[r][t], false, false);
                acc[r][t] = __builtin_amdgcn_wmma_f32_16x16x32_bf16(
                    false, a_hi[r], false, b_hi, (short)0, acc[r][t], false, false);
            }
        }
    }

    // ---- epilogue: C/D layout -> VGPR v holds M = v + 8*lhalf, N = l16 ----
#pragma unroll
    for (int r = 0; r < 2; ++r) {
#pragma unroll
        for (int t = 0; t < 4; ++t) {
            const int n  = n_base + nq * 64 + t * 16 + l16;
            const int m0 = m_base + r * 16 + lhalf * 8;
#pragma unroll
            for (int v = 0; v < 8; ++v) {
                size_t off = (size_t)(m0 + v) * N + n;
                float  val = acc[r][t][v];
                if (SPLIT) {
                    __bf16 h, l;
                    bf16_split(val, h, l);
                    Chi[off] = h;
                    Clo[off] = l;
                } else {
                    Cf32[off] = val;
                }
            }
        }
    }
}

// ---------------------------------------------------------------------------
// Launch: embed-pool -> 3 hashed GEMMs. Activations flow as hi/lo bf16 planes.
// Workspace (66 MB): pooled planes @ 0/1MB, h1 planes @ 2/18MB, h2 @ 34/50MB.
// ---------------------------------------------------------------------------
extern "C" void kernel_launch(void* const* d_in, const int* in_sizes, int n_in,
                              void* d_out, int out_size, void* d_ws, size_t ws_size,
                              hipStream_t stream) {
    const int*   x      = (const int*)d_in[0];
    const float* w      = (const float*)d_in[1];
    const float* hw_emb = (const float*)d_in[2];
    const float* hw0    = (const float*)d_in[3];
    const float* hw1    = (const float*)d_in[4];
    const float* hw2    = (const float*)d_in[5];
    float*       out    = (float*)d_out;

    char* ws = (char*)d_ws;
    __bf16* p_hi  = (__bf16*)(ws);                    // [2048, 256]  1MB
    __bf16* p_lo  = (__bf16*)(ws + (1ull  << 20));
    __bf16* h1_hi = (__bf16*)(ws + (2ull  << 20));    // [2048,4096] 16MB
    __bf16* h1_lo = (__bf16*)(ws + (18ull << 20));
    __bf16* h2_hi = (__bf16*)(ws + (34ull << 20));    // [2048,4096] 16MB
    __bf16* h2_lo = (__bf16*)(ws + (50ull << 20));

    const unsigned mask20 = (1u << 20) - 1u;
    const unsigned mask21 = (1u << 21) - 1u;

    // 1) pooled = embed+pool -> split planes       [2048, 256]
    rz_embed_pool_kernel<<<2048, 256, 0, stream>>>(x, w, hw_emb, p_hi, p_lo);

    // 2) h1 = pooled @ W0(hw0, seed 2)             [2048,256]x[256,4096]
    rz_gemm_bf16x3_kernel<true><<<dim3(4096 / 128, 2048 / 128), 256, 0, stream>>>(
        p_hi, p_lo, hw0, nullptr, h1_hi, h1_lo, 2048, 4096, 256, 2u, mask20);

    // 3) h2 = h1 @ W1(hw1, seed 3)                 [2048,4096]x[4096,4096]
    rz_gemm_bf16x3_kernel<true><<<dim3(4096 / 128, 2048 / 128), 256, 0, stream>>>(
        h1_hi, h1_lo, hw1, nullptr, h2_hi, h2_lo, 2048, 4096, 4096, 3u, mask21);

    // 4) out = h2 @ W2(hw2, seed 4) -> f32         [2048,4096]x[4096,2048]
    rz_gemm_bf16x3_kernel<false><<<dim3(2048 / 128, 2048 / 128), 256, 0, stream>>>(
        h2_hi, h2_lo, hw2, out, nullptr, nullptr, 2048, 2048, 4096, 4u, mask21);
}
